// OptimizedGNNFeatureSelector_15453292331117
// MI455X (gfx1250) — matrix operand order repaired and checked by reading, not compiled, not run on hardware
//
#include <hip/hip_runtime.h>
#include <hip/hip_bf16.h>
#include <math.h>

// ---------------------------------------------------------------------------
// Types for CDNA5 WMMA
// ---------------------------------------------------------------------------
typedef __attribute__((ext_vector_type(16))) _Float16 v16h;
typedef __attribute__((ext_vector_type(8)))  _Float16 v8h;
typedef __attribute__((ext_vector_type(8)))  float    v8f;

#define HIDW 64

// ---------------------------------------------------------------------------
// Tiled GEMM on f16 operands: C[M x NC] = A[M x K] * B[K x NC]
//   A  : row-major f16 [M x K]
//   BT : f16, B transposed -> [NC x K] (each output column's K-run contiguous)
//   C  : row-major f32
// One wave32 computes one 16x16 tile via V_WMMA_F32_16X16X32_F16.
// The 16xK A tile is staged in LDS once per block and shared by all waves
// (all waves in a block have the same tm). Fragment layouts per ISA 7.12.2:
//   A 16x32 : lanes 0-15 row M=lane, halves 0-7 -> K=0..7, 8-15 -> K=16..23
//             lanes 16-31 row M=lane-16,  halves -> K=8..15 / K=24..31
//   B 32x16 : lanes 0-15 col N=lane, halves 0-7 -> K=0..7, 8-15 -> K=16..23; lanes 16-31 -> K=8..15 / K=24..31
//   C/D     : lane n=lane&15, vgpr r -> M = (lane<16 ? r : 8+r)
// ---------------------------------------------------------------------------
template <int K, int NC>
__global__ void wmma_gemm_f16_kernel(const _Float16* __restrict__ A,
                                     const _Float16* __restrict__ BT,
                                     float* __restrict__ C) {
  constexpr int TILES_N = NC / 16;
  constexpr int WPB     = (TILES_N < 8) ? TILES_N : 8;   // waves per block
  constexpr int THREADS = WPB * 32;

  __shared__ _Float16 As[16 * K];                        // one A tile per block

  const int lane = threadIdx.x & 31;
  const int wave = threadIdx.x >> 5;
  const int tm   = blockIdx.x / (TILES_N / WPB);
  const int tng  = blockIdx.x % (TILES_N / WPB);
  const int tn   = tng * WPB + wave;

  // Cooperative stage of A tile: 16*K halves = 2*K uint4's (128-bit copies)
  {
    const uint4* __restrict__ src = (const uint4*)(A + (size_t)tm * 16 * K);
    uint4* dst = (uint4*)As;
    for (int i = threadIdx.x; i < 2 * K; i += THREADS) dst[i] = src[i];
  }
  __syncthreads();

  const int half_sel = (lane >> 4) & 1;        // 0: lanes 0-15, 1: lanes 16-31
  const int mrow     = lane & 15;              // row within tile (A), col (B/C)
  const int ncol     = tn * 16 + (lane & 15);

  v8f acc = {};
  for (int kk = 0; kk < K; kk += 32) {
    // A fragment: two contiguous 8-half runs from LDS (ds_load_b128)
    const v8h a0 = *(const v8h*)(As + mrow * K + kk + half_sel * 8);
    const v8h a1 = *(const v8h*)(As + mrow * K + kk + 16 + half_sel * 8);
    const v16h a = __builtin_shufflevector(a0, a1, 0, 1, 2, 3, 4, 5, 6, 7,
                                           8, 9, 10, 11, 12, 13, 14, 15);
    // B fragment: two 8-half runs (K=8h..8h+7, 16+8h..) from transposed B (2x global_load_b128)
    const _Float16* __restrict__ Bcol = BT + (size_t)ncol * K + kk;
    const v8h b0 = *(const v8h*)(Bcol + half_sel * 8);
    const v8h b1 = *(const v8h*)(Bcol + 16 + half_sel * 8);
    const v16h b = __builtin_shufflevector(b0, b1, 0, 1, 2, 3, 4, 5, 6, 7,
                                           8, 9, 10, 11, 12, 13, 14, 15);
    acc = __builtin_amdgcn_wmma_f32_16x16x32_f16(
        /*neg_a=*/false, a, /*neg_b=*/false, b,
        /*c_mod=*/(short)0, acc, /*reuse_a=*/false, /*reuse_b=*/false);
  }
#pragma unroll
  for (int r = 0; r < 8; ++r) {
    const int m = tm * 16 + half_sel * 8 + r;
    C[(size_t)m * NC + ncol] = acc[r];
  }
}

// ---------------------------------------------------------------------------
// BT[n*K + k] = (f16) B[k*NC + n]   (transpose + convert, tiny one-shot)
// ---------------------------------------------------------------------------
__global__ void transpose_f16_kernel(const float* __restrict__ B,
                                     _Float16* __restrict__ BT, int K, int NC) {
  const int i = blockIdx.x * blockDim.x + threadIdx.x;
  if (i >= K * NC) return;
  const int n = i / K, k = i % K;
  BT[i] = (_Float16)B[(size_t)k * NC + n];
}

// ---------------------------------------------------------------------------
// h = x @ W_in + b_in   (outer product per node), emitted directly in f16
// ---------------------------------------------------------------------------
__global__ void input_proj_kernel(const float* __restrict__ x,
                                  const float* __restrict__ W_in,
                                  const float* __restrict__ b_in,
                                  _Float16* __restrict__ hH, int n) {
  const int i = blockIdx.x * blockDim.x + threadIdx.x;
  if (i >= n * HIDW) return;
  const int node = i >> 6, c = i & 63;
  hH[i] = (_Float16)(x[node] * W_in[c] + b_in[c]);
}

// ---------------------------------------------------------------------------
// Per-(node, head) attention logits
// ---------------------------------------------------------------------------
__global__ void attn_scores_kernel(const float* __restrict__ hW,
                                   const float* __restrict__ a_src,
                                   const float* __restrict__ a_dst,
                                   float* __restrict__ as_, float* __restrict__ ad_,
                                   int n, int H) {
  const int i = blockIdx.x * blockDim.x + threadIdx.x;
  if (i >= n * H) return;
  const int node = i / H, hh = i % H;
  const float* __restrict__ row = hW + (size_t)node * (H * HIDW) + hh * HIDW;
  float s = 0.f, d = 0.f;
#pragma unroll 8
  for (int c = 0; c < HIDW; ++c) {
    s += row[c] * a_src[hh * HIDW + c];
    d += row[c] * a_dst[hh * HIDW + c];
  }
  as_[i] = s;
  ad_[i] = d;
}

// Order-preserving float <-> uint mapping for atomic max on floats.
__device__ __forceinline__ unsigned fenc(float f) {
  unsigned u = __float_as_uint(f);
  return (u & 0x80000000u) ? ~u : (u | 0x80000000u);
}
__device__ __forceinline__ float fdec(unsigned e) {
  unsigned u = (e & 0x80000000u) ? (e & 0x7FFFFFFFu) : ~e;
  return __uint_as_float(u);
}

__device__ __forceinline__ void edge_endpoints(const int* __restrict__ ei, int e,
                                               int E, int& si, int& di) {
  if (e < E) { si = ei[e]; di = ei[E + e]; }   // edge_index rows [2,E]
  else       { si = di = e - E; }              // appended self-loops
}

// segment_max over destinations (encoded-uint atomicMax)
__global__ void edge_max_kernel(const int* __restrict__ ei,
                                const float* __restrict__ as_,
                                const float* __restrict__ ad_,
                                unsigned* __restrict__ emax, int E, int n, int H) {
  const int e = blockIdx.x * blockDim.x + threadIdx.x;
  if (e >= E + n) return;
  int si, di; edge_endpoints(ei, e, E, si, di);
  for (int h = 0; h < H; ++h) {
    float v = as_[si * H + h] + ad_[di * H + h];
    v = (v < 0.f) ? 0.2f * v : v;              // leaky_relu(0.2)
    atomicMax(&emax[di * H + h], fenc(v));
  }
}

// denom = segment_sum(exp(e - emax[di]))
__global__ void edge_denom_kernel(const int* __restrict__ ei,
                                  const float* __restrict__ as_,
                                  const float* __restrict__ ad_,
                                  const unsigned* __restrict__ emax,
                                  float* __restrict__ denom, int E, int n, int H) {
  const int e = blockIdx.x * blockDim.x + threadIdx.x;
  if (e >= E + n) return;
  int si, di; edge_endpoints(ei, e, E, si, di);
  for (int h = 0; h < H; ++h) {
    float v = as_[si * H + h] + ad_[di * H + h];
    v = (v < 0.f) ? 0.2f * v : v;
    atomicAdd(&denom[di * H + h], __expf(v - fdec(emax[di * H + h])));
  }
}

// out[di] += alpha * hW[si]; one 64-thread group per edge, one lane per channel
__global__ void edge_aggr_kernel(const int* __restrict__ ei,
                                 const float* __restrict__ as_,
                                 const float* __restrict__ ad_,
                                 const unsigned* __restrict__ emax,
                                 const float* __restrict__ denom,
                                 const float* __restrict__ hW,
                                 float* __restrict__ out, int E, int n, int H) {
  const int gid = blockIdx.x * blockDim.x + threadIdx.x;
  const int e = gid >> 6;
  const int c = gid & 63;
  if (e >= E + n) return;
  int si, di; edge_endpoints(ei, e, E, si, di);
  for (int h = 0; h < H; ++h) {
    float v = as_[si * H + h] + ad_[di * H + h];
    v = (v < 0.f) ? 0.2f * v : v;
    const float ex    = __expf(v - fdec(emax[di * H + h]));
    const float alpha = ex / fmaxf(denom[di * H + h], 1e-16f);
    atomicAdd(&out[(size_t)di * (H * HIDW) + h * HIDW + c],
              alpha * hW[(size_t)si * (H * HIDW) + h * HIDW + c]);
  }
}

// bias+relu; writes f16 copy for the next WMMA stage, optional f32 copy
__global__ void bias_relu_kernel(const float* __restrict__ acc,
                                 const float* __restrict__ bias,
                                 _Float16* __restrict__ out16,
                                 float* __restrict__ out32,  // may be null
                                 int n, int width) {
  const int i = blockIdx.x * blockDim.x + threadIdx.x;
  if (i >= n * width) return;
  float v = acc[i] + bias[i % width];
  v = v > 0.f ? v : 0.f;
  out16[i] = (_Float16)v;
  if (out32) out32[i] = v;
}

// att = sigmoid(relu(t + fa_b1) @ fa_W2 + fa_b2)   (t = h2 @ fa_W1, from WMMA)
__global__ void feat_att_kernel(const float* __restrict__ t,
                                const float* __restrict__ fa_b1,
                                const float* __restrict__ fa_W2,
                                const float* __restrict__ fa_b2,
                                float* __restrict__ att, int n) {
  const int i = blockIdx.x * blockDim.x + threadIdx.x;
  if (i >= n) return;
  float s = fa_b2[0];
#pragma unroll 8
  for (int c = 0; c < HIDW; ++c) {
    float v = t[(size_t)i * HIDW + c] + fa_b1[c];
    v = v > 0.f ? v : 0.f;
    s += v * fa_W2[c];
  }
  att[i] = 1.f / (1.f + __expf(-s));
}

// pooled[b,f] = mean over the 64 contiguous nodes of graph b
__global__ void pool_kernel(const float* __restrict__ h2,
                            float* __restrict__ pooled, int nb) {
  const int i = blockIdx.x * blockDim.x + threadIdx.x;
  if (i >= nb * HIDW) return;
  const int b = i >> 6, f = i & 63;
  float s = 0.f;
#pragma unroll 8
  for (int k = 0; k < 64; ++k) s += h2[((size_t)b * 64 + k) * HIDW + f];
  pooled[i] = s * (1.f / 64.f);
}

// logits[b,c] = pooled[b] . cls_W[:,c] + cls_b[c]
__global__ void cls_kernel(const float* __restrict__ pooled,
                           const float* __restrict__ cls_W,
                           const float* __restrict__ cls_b,
                           float* __restrict__ out, int nb, int ncls) {
  const int i = blockIdx.x * blockDim.x + threadIdx.x;
  if (i >= nb * ncls) return;
  const int b = i / ncls, c = i % ncls;
  float s = cls_b[c];
#pragma unroll 8
  for (int f = 0; f < HIDW; ++f) s += pooled[(size_t)b * HIDW + f] * cls_W[f * ncls + c];
  out[i] = s;
}

// final_attention[f] = mean over graphs of att[b*64+f]
__global__ void final_att_kernel(const float* __restrict__ att,
                                 float* __restrict__ out, int nb) {
  const int f = threadIdx.x;
  if (f >= HIDW) return;
  float s = 0.f;
  for (int b = 0; b < nb; ++b) s += att[(size_t)b * 64 + f];
  out[f] = s / (float)nb;
}

// ---------------------------------------------------------------------------
// Host-side orchestration
// ---------------------------------------------------------------------------
extern "C" void kernel_launch(void* const* d_in, const int* in_sizes, int n_in,
                              void* d_out, int out_size, void* d_ws, size_t ws_size,
                              hipStream_t stream) {
  const float* x      = (const float*)d_in[0];
  const int*   ei     = (const int*)  d_in[1];
  // d_in[2] = batch (implicit: 64 contiguous nodes per graph)
  const float* W_in   = (const float*)d_in[3];
  const float* b_in   = (const float*)d_in[4];
  const float* W1     = (const float*)d_in[5];
  const float* a1_src = (const float*)d_in[6];
  const float* a1_dst = (const float*)d_in[7];
  const float* b1     = (const float*)d_in[8];
  const float* W2     = (const float*)d_in[9];
  const float* a2_src = (const float*)d_in[10];
  const float* a2_dst = (const float*)d_in[11];
  const float* b2     = (const float*)d_in[12];
  const float* fa_W1  = (const float*)d_in[13];
  const float* fa_b1  = (const float*)d_in[14];
  const float* fa_W2  = (const float*)d_in[15];
  const float* fa_b2  = (const float*)d_in[16];
  const float* cls_W  = (const float*)d_in[17];
  const float* cls_b  = (const float*)d_in[18];
  float* out = (float*)d_out;

  const int N    = in_sizes[0];        // 65536
  const int E    = in_sizes[1] / 2;    // 524288
  const int NB   = N / 64;             // 1024 graphs
  const int NCLS = in_sizes[18];       // 5
  const int H1   = 4;

  // Workspace carve-out (units of float; all sizes keep 16B alignment)
  float* ws = (float*)d_ws;
  size_t o = 0;
  auto carve = [&](size_t nf) { float* p = ws + o; o += nf; return p; };
  _Float16* hH     = (_Float16*)carve((size_t)N * 32);    // [N,64]  f16
  float*    hW1    = carve((size_t)N * 256);              // [N,256] f32
  float*    as1    = carve((size_t)N * 4);
  float*    ad1    = carve((size_t)N * 4);
  unsigned* emax1  = (unsigned*)carve((size_t)N * 4);
  float*    denom1 = carve((size_t)N * 4);
  float*    h1acc  = carve((size_t)N * 256);              // atomic target
  _Float16* h1H    = (_Float16*)carve((size_t)N * 128);   // [N,256] f16
  float*    hW2    = carve((size_t)N * 64);
  float*    as2    = carve((size_t)N);
  float*    ad2    = carve((size_t)N);
  unsigned* emax2  = (unsigned*)carve((size_t)N);
  float*    denom2 = carve((size_t)N);
  float*    h2acc  = carve((size_t)N * 64);               // atomic target
  float*    h2     = carve((size_t)N * 64);
  _Float16* h2H    = (_Float16*)carve((size_t)N * 32);    // [N,64]  f16
  float*    t      = carve((size_t)N * 64);
  float*    att    = carve((size_t)N);
  float*    pooled = carve((size_t)NB * 64);
  _Float16* W1T    = (_Float16*)carve((size_t)(256 * 64) / 2);  // [256,64]
  _Float16* W2T    = (_Float16*)carve((size_t)(64 * 256) / 2);  // [64,256]
  _Float16* faW1T  = (_Float16*)carve((size_t)(64 * 64) / 2);   // [64,64]
  (void)ws_size;

  const int TB = 256;
  const int EDG = E + N;
  auto cdiv = [](long long a, long long b) { return (unsigned)((a + b - 1) / b); };

  // Zero accumulators every launch (graph-capture-safe memset nodes)
  hipMemsetAsync(emax1,  0, (size_t)N * 4 * sizeof(unsigned), stream);
  hipMemsetAsync(denom1, 0, (size_t)N * 4 * sizeof(float),    stream);
  hipMemsetAsync(h1acc,  0, (size_t)N * 256 * sizeof(float),  stream);
  hipMemsetAsync(emax2,  0, (size_t)N * sizeof(unsigned),     stream);
  hipMemsetAsync(denom2, 0, (size_t)N * sizeof(float),        stream);
  hipMemsetAsync(h2acc,  0, (size_t)N * 64 * sizeof(float),   stream);

  // 0) weight conversion/transposition (tiny)
  transpose_f16_kernel<<<cdiv(64 * 256, TB), TB, 0, stream>>>(W1, W1T, 64, 256);
  transpose_f16_kernel<<<cdiv(256 * 64, TB), TB, 0, stream>>>(W2, W2T, 256, 64);
  transpose_f16_kernel<<<cdiv(64 * 64, TB), TB, 0, stream>>>(fa_W1, faW1T, 64, 64);

  // 1) input projection (f16 output)
  input_proj_kernel<<<cdiv((long long)N * 64, TB), TB, 0, stream>>>(x, W_in, b_in, hH, N);

  // 2) GAT1 linear: hW1 = h @ W1  [N,64]x[64,256]  (WMMA, 8 waves share A tile)
  wmma_gemm_f16_kernel<64, 256><<<(N / 16) * 2, 256, 0, stream>>>(hH, W1T, hW1);

  // 3) GAT1 attention + edge softmax + aggregation
  attn_scores_kernel<<<cdiv((long long)N * H1, TB), TB, 0, stream>>>(hW1, a1_src, a1_dst, as1, ad1, N, H1);
  edge_max_kernel  <<<cdiv(EDG, TB), TB, 0, stream>>>(ei, as1, ad1, emax1, E, N, H1);
  edge_denom_kernel<<<cdiv(EDG, TB), TB, 0, stream>>>(ei, as1, ad1, emax1, denom1, E, N, H1);
  edge_aggr_kernel <<<cdiv((long long)EDG * 64, TB), TB, 0, stream>>>(ei, as1, ad1, emax1, denom1, hW1, h1acc, E, N, H1);
  bias_relu_kernel <<<cdiv((long long)N * 256, TB), TB, 0, stream>>>(h1acc, b1, h1H, nullptr, N, 256);

  // 4) GAT2 linear: hW2 = h1 @ W2  [N,256]x[256,64]  (WMMA, 4 waves/block)
  wmma_gemm_f16_kernel<256, 64><<<(N / 16), 128, 0, stream>>>(h1H, W2T, hW2);

  // 5) GAT2 edge softmax (single head) + aggregation
  attn_scores_kernel<<<cdiv(N, TB), TB, 0, stream>>>(hW2, a2_src, a2_dst, as2, ad2, N, 1);
  edge_max_kernel  <<<cdiv(EDG, TB), TB, 0, stream>>>(ei, as2, ad2, emax2, E, N, 1);
  edge_denom_kernel<<<cdiv(EDG, TB), TB, 0, stream>>>(ei, as2, ad2, emax2, denom2, E, N, 1);
  edge_aggr_kernel <<<cdiv((long long)EDG * 64, TB), TB, 0, stream>>>(ei, as2, ad2, emax2, denom2, hW2, h2acc, E, N, 1);
  bias_relu_kernel <<<cdiv((long long)N * 64, TB), TB, 0, stream>>>(h2acc, b2, h2H, h2, N, 64);

  // 6) Feature-attention MLP: t = h2 @ fa_W1 (WMMA), fused relu/dot/sigmoid
  wmma_gemm_f16_kernel<64, 64><<<(N / 16), 128, 0, stream>>>(h2H, faW1T, t);
  feat_att_kernel<<<cdiv(N, TB), TB, 0, stream>>>(t, fa_b1, fa_W2, fa_b2, att, N);

  // 7) Per-graph mean pool + classifier -> out[0 .. NB*NCLS)
  pool_kernel<<<cdiv((long long)NB * 64, TB), TB, 0, stream>>>(h2, pooled, NB);
  cls_kernel <<<cdiv((long long)NB * NCLS, TB), TB, 0, stream>>>(pooled, cls_W, cls_b, out, NB, NCLS);

  // 8) final_attention -> out[NB*NCLS .. NB*NCLS+64)
  final_att_kernel<<<1, 64, 0, stream>>>(att, out + (size_t)NB * NCLS, NB);
}
